// MoE_58901181497482
// MI455X (gfx1250) — matrix operand order repaired and checked by reading, not compiled, not run on hardware
//
#include <hip/hip_runtime.h>
#include <math.h>

// ---------------- problem constants ----------------
#define TOK   8192      // B*S
#define HDIM  1024
#define NEXP  8
#define DFF   4096

// ---------------- types ----------------
typedef unsigned int u32;
typedef unsigned short u16;
typedef __attribute__((ext_vector_type(16))) __bf16 v16bf;
typedef __attribute__((ext_vector_type(8)))  float  v8f;
typedef __attribute__((ext_vector_type(4)))  u32    u32x4;

static __device__ __forceinline__ u16 f32_to_bf16(float f) {
    u32 u = __float_as_uint(f);
    u32 r = 0x7FFFu + ((u >> 16) & 1u);   // round-to-nearest-even
    return (u16)((u + r) >> 16);
}

static __device__ __forceinline__ v16bf load_frag(const u16* p0, const u16* p1) {
    union { u32x4 q[2]; v16bf v; } t;
    t.q[0] = *(const u32x4*)p0;   // 16B aligned
    t.q[1] = *(const u32x4*)p1;   // 16B aligned
    return t.v;
}

// ---------------- elementwise convert f32 -> bf16 ----------------
__global__ void convert_bf16_kernel(const float* __restrict__ src,
                                    u16* __restrict__ dst, int n) {
    int i = blockIdx.x * blockDim.x + threadIdx.x;
    if (i < n) dst[i] = f32_to_bf16(src[i]);
}

// ---------------- transpose + convert: src[R,C] f32 -> dst[C,R] bf16 ----------
__global__ __launch_bounds__(256)
void transpose_convert_kernel(const float* __restrict__ src,
                              u16* __restrict__ dst, int R, int C) {
    __shared__ float tile[32][33];
    const int c0 = blockIdx.x * 32;
    const int r0 = blockIdx.y * 32;
    const int tx = threadIdx.x;        // 0..31
    const int ty = threadIdx.y;        // 0..7
#pragma unroll
    for (int i = 0; i < 32; i += 8)
        tile[ty + i][tx] = src[(size_t)(r0 + ty + i) * C + c0 + tx];
    __syncthreads();
#pragma unroll
    for (int i = 0; i < 32; i += 8)
        dst[(size_t)(c0 + ty + i) * R + r0 + tx] = f32_to_bf16(tile[tx][ty + i]);
}

// ---------------- zero the exp_counts slots ----------------
__global__ void zero_counts_kernel(int* c) {
    if (threadIdx.x < NEXP) c[threadIdx.x] = 0;
}

// ---------------- gating: scores, top-2 histogram ----------------
__global__ __launch_bounds__(256)
void gate_kernel(const float* __restrict__ x, const float* __restrict__ gW,
                 const float* __restrict__ gb, float* __restrict__ gs,
                 int* __restrict__ counts) {
    const int t    = blockIdx.x;
    const int lane = threadIdx.x & 31;
    const int e    = threadIdx.x >> 5;          // wave -> expert (8 waves)
    const float* xr = x + (size_t)t * HDIM;
    float acc = 0.f;
    for (int i = lane; i < HDIM; i += 32)
        acc += xr[i] * gW[(size_t)i * NEXP + e];
#pragma unroll
    for (int off = 16; off > 0; off >>= 1)
        acc += __shfl_down(acc, off, 32);
    __shared__ float sc[NEXP];
    if (lane == 0) {
        float s = acc + gb[e];
        sc[e] = s;
        gs[(size_t)t * NEXP + e] = s;
    }
    __syncthreads();
    if (threadIdx.x == 0) {
        int i1 = 0;
#pragma unroll
        for (int k = 1; k < NEXP; k++) if (sc[k] > sc[i1]) i1 = k;
        int i2 = (i1 == 0) ? 1 : 0;
#pragma unroll
        for (int k = 0; k < NEXP; k++) {
            if (k == i1) continue;
            if (sc[k] > sc[i2]) i2 = k;
        }
        atomicAdd(&counts[i1], 1);
        atomicAdd(&counts[i2], 1);
    }
}

// ---------------- aux loss: sum_e (mean_t gs[t,e])^2, deterministic ----------
__global__ __launch_bounds__(256)
void aux_kernel(const float* __restrict__ gs, float* __restrict__ aux_out) {
    const int lane = threadIdx.x & 31;
    const int e    = threadIdx.x >> 5;
    float acc = 0.f;
    for (int t = lane; t < TOK; t += 32)
        acc += gs[(size_t)t * NEXP + e];
#pragma unroll
    for (int off = 16; off > 0; off >>= 1)
        acc += __shfl_down(acc, off, 32);
    __shared__ float cs[NEXP];
    if (lane == 0) cs[e] = acc;
    __syncthreads();
    if (threadIdx.x == 0) {
        float a = 0.f;
#pragma unroll
        for (int k = 0; k < NEXP; k++) {
            float m = cs[k] * (1.0f / (float)TOK);
            a += m * m;
        }
        aux_out[0] = a;
    }
}

// ---------------- bf16 WMMA GEMM: C[M,N] = A[M,K] * BT[N,K]^T + bias --------
// BT is stored [N,K] row-major (K-contiguous per output column) so both WMMA
// operand fragments are contiguous 16B LDS loads. LDS is ping-pong double
// buffered: next tile's global loads issue before the WMMA block, one barrier
// per K-step.
#define BM 128
#define BN 128
#define BK 32
#define LSTR 40   // padded LDS row stride in halves (80B) -> conflict-light

template <int GELU>
__global__ __launch_bounds__(256)
void gemm_bf16_kernel(const u16* __restrict__ A, const u16* __restrict__ BT,
                      const float* __restrict__ bias,
                      float* __restrict__ outF, u16* __restrict__ outB,
                      int M, int N, int K) {
    __shared__ __align__(16) u16 As[2][BM * LSTR];
    __shared__ __align__(16) u16 Bs[2][BN * LSTR];

    const int tid  = threadIdx.x;
    const int lane = tid & 31;
    const int wave = tid >> 5;       // 0..7
    const int wm   = wave & 3;       // 4 waves along M (32 rows each)
    const int wn   = wave >> 2;      // 2 waves along N (64 cols each)
    const int m0   = blockIdx.y * BM;
    const int n0   = blockIdx.x * BN;

    const int r  = lane & 15;
    const int hi = lane >> 4;

    v8f acc[2][4];
#pragma unroll
    for (int mt = 0; mt < 2; mt++)
#pragma unroll
        for (int nt = 0; nt < 4; nt++)
            acc[mt][nt] = (v8f)(0.0f);

    const int crow = tid >> 1;            // 0..127
    const int cseg = (tid & 1) * 16;      // halves
    const u16* gA = A  + (size_t)(m0 + crow) * K + cseg;
    const u16* gB = BT + (size_t)(n0 + crow) * K + cseg;
    u16* const lA = &As[0][0] + crow * LSTR + cseg;
    u16* const lB = &Bs[0][0] + crow * LSTR + cseg;

    const int nsteps = K / BK;

    // prologue: stage tile 0
    {
        u32x4 va = *(const u32x4*)(gA);
        u32x4 vb = *(const u32x4*)(gB);
        *(u32x4*)(lA) = va;
        *(u32x4*)(lB) = vb;
    }
    __syncthreads();

    int cur = 0;
    for (int s = 0; s < nsteps; s++) {
        // issue next tile's global loads before compute (latency overlap)
        u32x4 va, vb;
        const bool have_next = (s + 1 < nsteps);
        if (have_next) {
            va = *(const u32x4*)(gA + (size_t)(s + 1) * BK);
            vb = *(const u32x4*)(gB + (size_t)(s + 1) * BK);
        }

        const u16* as = &As[cur][0];
        const u16* bs = &Bs[cur][0];
        v16bf afrag[2], bfrag[4];
#pragma unroll
        for (int mt = 0; mt < 2; mt++) {
            const u16* base = as + (wm * 32 + mt * 16 + r) * LSTR + hi * 8;
            afrag[mt] = load_frag(base, base + 16);    // K 0-7/8-15 then +16
        }
#pragma unroll
        for (int nt = 0; nt < 4; nt++) {
            const u16* base = bs + (wn * 64 + nt * 16 + r) * LSTR + hi * 8;
            bfrag[nt] = load_frag(base, base + 16);    // K 0-7/8-15 then +16
        }
#pragma unroll
        for (int mt = 0; mt < 2; mt++)
#pragma unroll
            for (int nt = 0; nt < 4; nt++)
                acc[mt][nt] = __builtin_amdgcn_wmma_f32_16x16x32_bf16(
                    false, afrag[mt], false, bfrag[nt],
                    (short)0, acc[mt][nt], false, false);

        if (have_next) {
            const int nxt = cur ^ 1;
            *(u32x4*)(&As[nxt][0] + crow * LSTR + cseg) = va;
            *(u32x4*)(&Bs[nxt][0] + crow * LSTR + cseg) = vb;
        }
        __syncthreads();
        cur ^= 1;
    }

    // epilogue: C layout VGPR rr -> (M = hi*8 + rr, N = lane&15)
#pragma unroll
    for (int mt = 0; mt < 2; mt++) {
#pragma unroll
        for (int nt = 0; nt < 4; nt++) {
            const int col = n0 + wn * 64 + nt * 16 + r;
            const float bc = bias[col];
#pragma unroll
            for (int rr = 0; rr < 8; rr++) {
                const int row = m0 + wm * 32 + mt * 16 + hi * 8 + rr;
                float v = acc[mt][nt][rr] + bc;
                if (GELU) {
                    float g = 0.5f * v * (1.0f + erff(v * 0.70710678118654752f));
                    outB[(size_t)row * N + col] = f32_to_bf16(g);
                } else {
                    outF[(size_t)row * N + col] = v;
                }
            }
        }
    }
}

// ---------------- host entry ----------------
extern "C" void kernel_launch(void* const* d_in, const int* in_sizes, int n_in,
                              void* d_out, int out_size, void* d_ws, size_t ws_size,
                              hipStream_t stream) {
    (void)in_sizes; (void)n_in; (void)out_size; (void)ws_size;

    const float* x  = (const float*)d_in[0];   // [TOK, HDIM]
    const float* gW = (const float*)d_in[1];   // [HDIM, NEXP]
    const float* gb = (const float*)d_in[2];   // [NEXP]
    const float* W1 = (const float*)d_in[3];   // [HDIM, DFF]
    const float* b1 = (const float*)d_in[4];   // [DFF]
    const float* W2 = (const float*)d_in[5];   // [DFF, HDIM]
    const float* b2 = (const float*)d_in[6];   // [HDIM]

    float* outF   = (float*)d_out;                         // [TOK*HDIM]
    float* auxF   = outF + (size_t)TOK * HDIM;             // [1]
    int*   counts = (int*)(outF + (size_t)TOK * HDIM + 1); // [NEXP] int32 bits

    char* ws = (char*)d_ws;
    size_t off = 0;
    u16* xb   = (u16*)(ws + off); off += (size_t)TOK * HDIM * 2;   // 16 MB
    u16* w1t  = (u16*)(ws + off); off += (size_t)DFF * HDIM * 2;   // 8 MB  [DFF,HDIM]
    u16* w2t  = (u16*)(ws + off); off += (size_t)HDIM * DFF * 2;   // 8 MB  [HDIM,DFF]
    u16* hddb = (u16*)(ws + off); off += (size_t)TOK * DFF * 2;    // 64 MB [TOK,DFF]
    float* gs = (float*)(ws + off);                                 // [TOK,NEXP]

    zero_counts_kernel<<<1, 32, 0, stream>>>(counts);

    convert_bf16_kernel<<<(TOK * HDIM + 255) / 256, 256, 0, stream>>>(
        x, xb, TOK * HDIM);
    transpose_convert_kernel<<<dim3(DFF / 32, HDIM / 32), dim3(32, 8), 0, stream>>>(
        W1, w1t, HDIM, DFF);   // W1[H,DFF] -> w1t[DFF,H]
    transpose_convert_kernel<<<dim3(HDIM / 32, DFF / 32), dim3(32, 8), 0, stream>>>(
        W2, w2t, DFF, HDIM);   // W2[DFF,H] -> w2t[H,DFF]

    gate_kernel<<<TOK, 256, 0, stream>>>(x, gW, gb, gs, counts);
    aux_kernel<<<1, 256, 0, stream>>>(gs, auxF);

    // hdd = gelu(x @ W1 + b1)  -> bf16
    gemm_bf16_kernel<1><<<dim3(DFF / BN, TOK / BM), 256, 0, stream>>>(
        xb, w1t, b1, nullptr, hddb, TOK, DFF, HDIM);
    // out = hdd @ W2 + b2      -> f32
    gemm_bf16_kernel<0><<<dim3(HDIM / BN, TOK / BM), 256, 0, stream>>>(
        hddb, w2t, b2, outF, nullptr, TOK, HDIM, DFF);
}
